// MoE_62878321214273
// MI455X (gfx1250) — matrix operand order repaired and checked by reading, not compiled, not run on hardware
//
#include <hip/hip_runtime.h>
#include <hip/hip_bf16.h>

typedef __attribute__((ext_vector_type(16))) __bf16 v16bf;
typedef __attribute__((ext_vector_type(8)))  __bf16 v8bf;
typedef __attribute__((ext_vector_type(8)))  float  v8f;
typedef __attribute__((ext_vector_type(4)))  unsigned int v4u;
typedef __attribute__((ext_vector_type(8)))  int v8i;
typedef __attribute__((ext_vector_type(4)))  int v4i;

#define DIMD   1024
#define NEXP   8
#define TOKS   4096
#define FDIM   4096
#define CAP    640
#define MAXN   1280          // 2*CAP tokens max per expert
#define MTILES 10            // MAXN/128

// ---------------------------------------------------------------- router ----
__global__ __launch_bounds__(256)
void moe_router_kernel(const float* __restrict__ x, const float* __restrict__ Wr,
                       int* __restrict__ topi, float* __restrict__ topv)
{
    const int lane = threadIdx.x & 31;
    const int wv   = threadIdx.x >> 5;
    const int t    = blockIdx.x * 8 + wv;
    if (t >= TOKS) return;

    float acc[NEXP];
#pragma unroll
    for (int e = 0; e < NEXP; ++e) acc[e] = 0.f;

    const float* xr = x + (size_t)t * DIMD;
    for (int d = lane; d < DIMD; d += 32) {
        float xv = xr[d];
        const float4* wr = (const float4*)(Wr + (size_t)d * NEXP);
        float4 w0 = wr[0], w1 = wr[1];
        acc[0] += xv * w0.x; acc[1] += xv * w0.y;
        acc[2] += xv * w0.z; acc[3] += xv * w0.w;
        acc[4] += xv * w1.x; acc[5] += xv * w1.y;
        acc[6] += xv * w1.z; acc[7] += xv * w1.w;
    }
#pragma unroll
    for (int e = 0; e < NEXP; ++e) {
#pragma unroll
        for (int off = 16; off > 0; off >>= 1)
            acc[e] += __shfl_xor(acc[e], off, 32);
    }
    if (lane == 0) {
        int i0 = 0;
#pragma unroll
        for (int e = 1; e < NEXP; ++e) if (acc[e] > acc[i0]) i0 = e;  // lowest-index ties
        int i1 = (i0 == 0) ? 1 : 0;
#pragma unroll
        for (int e = 0; e < NEXP; ++e)
            if (e != i0 && acc[e] > acc[i1]) i1 = e;
        // renormalized top-2 softmax gates: exp(l)/sum over {l0,l1}
        float p1  = __expf(acc[i1] - acc[i0]);
        float inv = 1.f / (1.f + p1);
        topi[2*t + 0] = i0;      topi[2*t + 1] = i1;
        topv[2*t + 0] = inv;     topv[2*t + 1] = p1 * inv;
    }
}

// ------------------------------------------------------ capacity assignment ----
__global__ void moe_assign_kernel(const int* __restrict__ topi, const float* __restrict__ topv,
                                  int* __restrict__ tok_list, float* __restrict__ gate_list,
                                  int* __restrict__ counts, unsigned char* __restrict__ kept)
{
    __shared__ int kc[2][NEXP];
    const int j = threadIdx.x;
    if (j < 16) {
        const int k = j >> 3, e = j & 7;
        int c = 0;
        for (int t = 0; t < TOKS; ++t)
            if (topi[2*t + k] == e) ++c;
        kc[k][e] = c < CAP ? c : CAP;
    }
    __syncthreads();
    if (j < 16) {
        const int k = j >> 3, e = j & 7;
        const int base = (k == 1) ? kc[0][e] : 0;
        if (k == 1) counts[e] = kc[0][e] + kc[1][e];
        int c = 0;
        for (int t = 0; t < TOKS; ++t) {
            if (topi[2*t + k] == e) {
                ++c;
                const bool kp = (c <= CAP);          // rank in flattened token order
                kept[2*t + k] = kp ? 1 : 0;
                if (kp) {
                    const int idx = base + c - 1;
                    tok_list [e*MAXN + idx] = 2*t + k;
                    gate_list[e*MAXN + idx] = topv[2*t + k];
                }
            }
        }
    }
}

// ------------------------------------------------------------------- GEMM1 ----
// h[e, m, n] = gelu( sum_k x[tok(m), k] * W1[e, k, n] + b1[e, n] )   (bf16 out)
__global__ __launch_bounds__(256)
void moe_gemm1_kernel(const float* __restrict__ x, const float* __restrict__ W1,
                      const float* __restrict__ b1, const int* __restrict__ tok_list,
                      const int* __restrict__ counts, __bf16* __restrict__ hbuf)
{
    const int e   = blockIdx.z;
    const int cnt = counts[e];
    const int m0  = blockIdx.y * 128;
    if (m0 >= cnt) return;
    const int n0  = blockIdx.x * 128;

    __shared__ __bf16 As[128 * 32];   // [m][k]
    __shared__ __bf16 Bs[128 * 32];   // [n][k] (transposed during staging)
    __shared__ int    rowTok[128];

    const int tid = threadIdx.x;
    if (tid < 128) {
        const int r = m0 + tid;
        rowTok[tid] = (r < cnt) ? (tok_list[e*MAXN + r] >> 1) : -1;
    }
    __syncthreads();

    const int lane = tid & 31;
    const int wv   = tid >> 5;
    const int wm   = wv >> 1;                  // 4 waves in M -> 32 rows each
    const int wn   = wv & 1;                   // 2 waves in N -> 64 cols each
    const int mrow = lane & 15;
    const int kbA  = (lane >> 4) * 8;          // A frag K-half base
    const int kbB  = (lane >> 4) * 8;          // B frag K-half base

    v8f acc[2][4];
#pragma unroll
    for (int im = 0; im < 2; ++im)
#pragma unroll
        for (int jn = 0; jn < 4; ++jn) acc[im][jn] = (v8f)0.f;

    for (int k0 = 0; k0 < DIMD; k0 += 32) {
        // stage A: 128 rows x 32 K, fp32 -> bf16, gathered token rows
        {
            const int r    = tid >> 1;
            const int cseg = (tid & 1) * 16;
            const int tok  = rowTok[r];
            float v[16];
            if (tok >= 0) {
                const float4* p = (const float4*)(x + (size_t)tok * DIMD + k0 + cseg);
#pragma unroll
                for (int q = 0; q < 4; ++q) {
                    float4 f = p[q];
                    v[4*q+0] = f.x; v[4*q+1] = f.y; v[4*q+2] = f.z; v[4*q+3] = f.w;
                }
            } else {
#pragma unroll
                for (int q = 0; q < 16; ++q) v[q] = 0.f;
            }
#pragma unroll
            for (int q = 0; q < 16; ++q) As[r*32 + cseg + q] = (__bf16)v[q];
        }
        // stage B: W1 tile [32 k][128 n] -> LDS transposed [n][k], fp32 -> bf16
        {
            const int kr   = tid >> 3;
            const int nseg = (tid & 7) * 16;
            const float4* p = (const float4*)(W1 + ((size_t)e * DIMD + (k0 + kr)) * FDIM + n0 + nseg);
#pragma unroll
            for (int q = 0; q < 4; ++q) {
                float4 f = p[q];
                Bs[(nseg + 4*q + 0)*32 + kr] = (__bf16)f.x;
                Bs[(nseg + 4*q + 1)*32 + kr] = (__bf16)f.y;
                Bs[(nseg + 4*q + 2)*32 + kr] = (__bf16)f.z;
                Bs[(nseg + 4*q + 3)*32 + kr] = (__bf16)f.w;
            }
        }
        __syncthreads();

        union { v16bf v; v8bf p[2]; } afr[2];
#pragma unroll
        for (int im = 0; im < 2; ++im) {
            const v8bf* ap = (const v8bf*)(As + (wm*32 + im*16 + mrow)*32 + kbA);
            afr[im].p[0] = ap[0];      // K = kb..kb+7
            afr[im].p[1] = ap[2];      // K = kb+16..kb+23
        }
#pragma unroll
        for (int jn = 0; jn < 4; ++jn) {
            union { v16bf v; v8bf p[2]; } bfr;
            const v8bf* bp = (const v8bf*)(Bs + (wn*64 + jn*16 + mrow)*32 + kbB);
            bfr.p[0] = bp[0];
            bfr.p[1] = bp[2];
            acc[0][jn] = __builtin_amdgcn_wmma_f32_16x16x32_bf16(
                false, afr[0].v, false, bfr.v, (short)0, acc[0][jn], false, false);
            acc[1][jn] = __builtin_amdgcn_wmma_f32_16x16x32_bf16(
                false, afr[1].v, false, bfr.v, (short)0, acc[1][jn], false, false);
        }
        __syncthreads();
    }

    // epilogue: +b1, exact GELU, store bf16 h
    const bool hihalf = (lane >= 16);
#pragma unroll
    for (int im = 0; im < 2; ++im) {
#pragma unroll
        for (int jn = 0; jn < 4; ++jn) {
#pragma unroll
            for (int i = 0; i < 8; ++i) {
                const int m = m0 + wm*32 + im*16 + i + (hihalf ? 8 : 0);
                if (m < cnt) {
                    const int n = n0 + wn*64 + jn*16 + mrow;
                    float v = acc[im][jn][i] + b1[e*FDIM + n];
                    float g = 0.5f * v * (1.f + erff(v * 0.70710678118654752f));
                    hbuf[((size_t)e * MAXN + m) * FDIM + n] = (__bf16)g;
                }
            }
        }
    }
}

// ------------------------------------------------------------------- GEMM2 ----
// buf[slot(m), n] = gate[m] * ( sum_k h[e, m, k] * W2[e, k, n] + b2[e, n] )
// A tile (bf16, contiguous rows) is fetched by the Tensor Data Mover.
__global__ __launch_bounds__(256)
void moe_gemm2_kernel(const __bf16* __restrict__ hbuf, const float* __restrict__ W2,
                      const float* __restrict__ b2, const int* __restrict__ tok_list,
                      const float* __restrict__ gate_list, const int* __restrict__ counts,
                      float* __restrict__ buf)
{
    const int e   = blockIdx.z;
    const int cnt = counts[e];
    const int m0  = blockIdx.y * 128;
    if (m0 >= cnt) return;
    const int n0  = blockIdx.x * 128;

    __shared__ __bf16 As[128 * 32];
    __shared__ __bf16 Bs[128 * 32];

    const int tid  = threadIdx.x;
    const int lane = tid & 31;
    const int wv   = tid >> 5;
    const int wm   = wv >> 1;
    const int wn   = wv & 1;
    const int mrow = lane & 15;
    const int kbA  = (lane >> 4) * 8;
    const int kbB  = (lane >> 4) * 8;

    // TDM descriptor constants (uniform)
    const unsigned ldsA = (unsigned)(size_t)(void*)As;                 // LDS byte offset
    const unsigned long long gaBase =
        (unsigned long long)(size_t)(const void*)(hbuf + ((size_t)e * MAXN + m0) * FDIM);
    const unsigned rowsAvail = (unsigned)(cnt - m0);                   // OOB rows read as 0

    v8f acc[2][4];
#pragma unroll
    for (int im = 0; im < 2; ++im)
#pragma unroll
        for (int jn = 0; jn < 4; ++jn) acc[im][jn] = (v8f)0.f;

    for (int k0 = 0; k0 < FDIM; k0 += 32) {
        // stage A via Tensor Data Mover: 128 x 32 bf16 tile, rows packed to [r][32]
        if (tid < 32) {
            const unsigned long long ga = gaBase + (unsigned long long)k0 * 2ull;
            v4u g0 = { 1u,                                   // count=1 descriptor
                       ldsA,                                 // lds_addr (bytes)
                       (unsigned)(ga & 0xFFFFFFFFull),
                       (unsigned)((ga >> 32) & 0x01FFFFFFull) | (2u << 30) };  // type=2
            const unsigned td0 = (unsigned)(FDIM - k0);      // remaining elems in row
            const unsigned td1 = rowsAvail;                  // remaining valid rows
            v8i g1 = { (int)(1u << 16),                      // data_size = 2 bytes
                       (int)((td0 & 0xFFFFu) << 16),         // tensor_dim0 lo
                       (int)(((td0 >> 16) & 0xFFFFu) | ((td1 & 0xFFFFu) << 16)),
                       (int)(((td1 >> 16) & 0xFFFFu) | (32u << 16)),   // tile_dim0 = 32
                       (int)128,                             // tile_dim1 = 128 rows
                       (int)FDIM,                            // tensor_dim0_stride lo
                       0, 0 };
            v4i gz = { 0, 0, 0, 0 };
#if defined(__clang_major__) && __clang_major__ >= 23
            v8i gz8 = { 0, 0, 0, 0, 0, 0, 0, 0 };
            __builtin_amdgcn_tensor_load_to_lds(g0, g1, gz, gz, gz8, 0);
#else
            __builtin_amdgcn_tensor_load_to_lds(g0, g1, gz, gz, 0);
#endif
        }
        // stage B: W2 tile [32 k][128 n] -> LDS transposed [n][k], fp32 -> bf16
        {
            const int kr   = tid >> 3;
            const int nseg = (tid & 7) * 16;
            const float4* p = (const float4*)(W2 + ((size_t)e * FDIM + (k0 + kr)) * DIMD + n0 + nseg);
#pragma unroll
            for (int q = 0; q < 4; ++q) {
                float4 f = p[q];
                Bs[(nseg + 4*q + 0)*32 + kr] = (__bf16)f.x;
                Bs[(nseg + 4*q + 1)*32 + kr] = (__bf16)f.y;
                Bs[(nseg + 4*q + 2)*32 + kr] = (__bf16)f.z;
                Bs[(nseg + 4*q + 3)*32 + kr] = (__bf16)f.w;
            }
        }
        if (tid < 32) __builtin_amdgcn_s_wait_tensorcnt(0);
        __syncthreads();

        union { v16bf v; v8bf p[2]; } afr[2];
#pragma unroll
        for (int im = 0; im < 2; ++im) {
            const v8bf* ap = (const v8bf*)(As + (wm*32 + im*16 + mrow)*32 + kbA);
            afr[im].p[0] = ap[0];
            afr[im].p[1] = ap[2];
        }
#pragma unroll
        for (int jn = 0; jn < 4; ++jn) {
            union { v16bf v; v8bf p[2]; } bfr;
            const v8bf* bp = (const v8bf*)(Bs + (wn*64 + jn*16 + mrow)*32 + kbB);
            bfr.p[0] = bp[0];
            bfr.p[1] = bp[2];
            acc[0][jn] = __builtin_amdgcn_wmma_f32_16x16x32_bf16(
                false, afr[0].v, false, bfr.v, (short)0, acc[0][jn], false, false);
            acc[1][jn] = __builtin_amdgcn_wmma_f32_16x16x32_bf16(
                false, afr[1].v, false, bfr.v, (short)0, acc[1][jn], false, false);
        }
        __syncthreads();
    }

    // epilogue: +b2, gate scale, scatter to disjoint (token, slot) rows
    const bool hihalf = (lane >= 16);
    int   slot[2][8];
    float gate[2][8];
#pragma unroll
    for (int im = 0; im < 2; ++im) {
#pragma unroll
        for (int i = 0; i < 8; ++i) {
            const int m = m0 + wm*32 + im*16 + i + (hihalf ? 8 : 0);
            if (m < cnt) { slot[im][i] = tok_list[e*MAXN + m]; gate[im][i] = gate_list[e*MAXN + m]; }
            else         { slot[im][i] = -1;                   gate[im][i] = 0.f; }
        }
    }
#pragma unroll
    for (int im = 0; im < 2; ++im) {
#pragma unroll
        for (int jn = 0; jn < 4; ++jn) {
#pragma unroll
            for (int i = 0; i < 8; ++i) {
                if (slot[im][i] >= 0) {
                    const int n = n0 + wn*64 + jn*16 + mrow;
                    buf[(size_t)slot[im][i] * DIMD + n] =
                        gate[im][i] * (acc[im][jn][i] + b2[e*DIMD + n]);
                }
            }
        }
    }
}

// ----------------------------------------------------------------- combine ----
__global__ __launch_bounds__(256)
void moe_combine_kernel(const float* __restrict__ buf, const unsigned char* __restrict__ kept,
                        float* __restrict__ out)
{
    const int idx = blockIdx.x * 256 + threadIdx.x;
    if (idx >= TOKS * DIMD) return;
    const int t = idx >> 10;
    const int d = idx & (DIMD - 1);
    float a = kept[2*t + 0] ? buf[(size_t)(2*t + 0) * DIMD + d] : 0.f;
    float b = kept[2*t + 1] ? buf[(size_t)(2*t + 1) * DIMD + d] : 0.f;
    out[idx] = a + b;
}

// ------------------------------------------------------------------ launch ----
extern "C" void kernel_launch(void* const* d_in, const int* in_sizes, int n_in,
                              void* d_out, int out_size, void* d_ws, size_t ws_size,
                              hipStream_t stream)
{
    const float* x  = (const float*)d_in[0];
    const float* Wr = (const float*)d_in[1];
    const float* W1 = (const float*)d_in[2];
    const float* b1 = (const float*)d_in[3];
    const float* W2 = (const float*)d_in[4];
    const float* b2 = (const float*)d_in[5];
    float* out = (float*)d_out;

    char* w = (char*)d_ws;
    constexpr size_t OFF_TOPI = 0;
    constexpr size_t OFF_TOPV = OFF_TOPI + (size_t)TOKS * 2 * 4;
    constexpr size_t OFF_TOK  = OFF_TOPV + (size_t)TOKS * 2 * 4;
    constexpr size_t OFF_GATE = OFF_TOK  + (size_t)NEXP * MAXN * 4;
    constexpr size_t OFF_CNT  = OFF_GATE + (size_t)NEXP * MAXN * 4;
    constexpr size_t OFF_KEPT = OFF_CNT  + 256;
    constexpr size_t OFF_BUF  = OFF_KEPT + ((size_t)TOKS * 2 + 255) / 256 * 256;
    constexpr size_t OFF_H    = OFF_BUF  + (size_t)TOKS * 2 * DIMD * 4;

    int*            topi      = (int*)           (w + OFF_TOPI);
    float*          topv      = (float*)         (w + OFF_TOPV);
    int*            tok_list  = (int*)           (w + OFF_TOK);
    float*          gate_list = (float*)         (w + OFF_GATE);
    int*            counts    = (int*)           (w + OFF_CNT);
    unsigned char*  kept      = (unsigned char*) (w + OFF_KEPT);
    float*          buf       = (float*)         (w + OFF_BUF);
    __bf16*         hbuf      = (__bf16*)        (w + OFF_H);

    moe_router_kernel<<<TOKS / 8, 256, 0, stream>>>(x, Wr, topi, topv);
    moe_assign_kernel<<<1, 32, 0, stream>>>(topi, topv, tok_list, gate_list, counts, kept);
    moe_gemm1_kernel<<<dim3(FDIM / 128, MTILES, NEXP), 256, 0, stream>>>(
        x, W1, b1, tok_list, counts, hbuf);
    moe_gemm2_kernel<<<dim3(DIMD / 128, MTILES, NEXP), 256, 0, stream>>>(
        hbuf, W2, b2, tok_list, gate_list, counts, buf);
    moe_combine_kernel<<<(TOKS * DIMD + 255) / 256, 256, 0, stream>>>(buf, kept, out);
}
